// GNNLayer_49331994361926
// MI455X (gfx1250) — matrix operand order repaired and checked by reading, not compiled, not run on hardware
//
#include <hip/hip_runtime.h>

typedef __attribute__((ext_vector_type(16))) _Float16 v16h;
typedef __attribute__((ext_vector_type(8)))  _Float16 v8h;
typedef __attribute__((ext_vector_type(8)))  float    v8f;

#define IN_DIM   256
#define OUT_DIM  128
#define NNODES   4096
#define NBATCH   8
#define TILES16_PER_BATCH (NNODES / 16)   // 256
#define TILES32_PER_BATCH (NNODES / 32)   // 128
#define WAVES_PER_BLOCK 4
#define LOG2E 1.44269504f

// ---------------------------------------------------------------------------
// Kernel 0: WT[n][k] = (f16) W[k][n]   (128 x 256)
// ---------------------------------------------------------------------------
__global__ void wt_convert_kernel(const float* __restrict__ W,
                                  _Float16* __restrict__ WT) {
    int idx = blockIdx.x * 256 + threadIdx.x;   // 0 .. 32767
    int n = idx >> 8;
    int k = idx & 255;
    WT[n * IN_DIM + k] = (_Float16)W[(size_t)k * OUT_DIM + n];
}

// ---------------------------------------------------------------------------
// Kernel 1: h = x@W + b  via v_wmma_f32_16x16x32_f16. Per 16-row tile emits:
//   hT[b][n][k]           f16 transposed h (phase-2 B matrix)
//   e1[b][j] = exp2(-log2e*(h_j . w1 + att_b))     } factored sigmoid
//   e2[b][i] = exp2(-log2e*(h_i . w2))             } terms
// ---------------------------------------------------------------------------
__global__ __launch_bounds__(32 * WAVES_PER_BLOCK)
void gat_h_kernel(const float*    __restrict__ x,
                  const _Float16* __restrict__ WT,
                  const float*    __restrict__ bias,
                  const float*    __restrict__ att_w,
                  const float*    __restrict__ att_b,
                  _Float16*       __restrict__ hT,
                  float*          __restrict__ e1o,
                  float*          __restrict__ e2o) {
    __shared__ float lds[WAVES_PER_BLOCK * 16 * OUT_DIM];

    const int lane  = threadIdx.x & 31;
    const int wave  = threadIdx.x >> 5;
    const int tile  = blockIdx.x * WAVES_PER_BLOCK + wave;  // 0..2047
    const int b     = tile >> 8;
    const int rt    = tile & (TILES16_PER_BATCH - 1);
    const int mlo   = lane & 15;
    const int hi    = lane >> 4;
    const int kbase = hi * 8;
    const int kofs  = hi * 16;
    const int moff  = hi * 8;

    v8f acc[8] = {};

    const float* xrow = x + (size_t)(b * NNODES + rt * 16 + mlo) * IN_DIM;

    #pragma unroll
    for (int kk = 0; kk < IN_DIM; kk += 32) {
        v8f xa = *(const v8f*)(xrow + kk + kbase);
        v8f xb = *(const v8f*)(xrow + kk + 16 + kbase);
        v16h a;
        #pragma unroll
        for (int i = 0; i < 8; ++i) {
            a[i]     = (_Float16)xa[i];
            a[i + 8] = (_Float16)xb[i];
        }
        #pragma unroll
        for (int t = 0; t < 8; ++t) {
            const _Float16* wp = WT + (size_t)(t * 16 + mlo) * IN_DIM + kk;
            const v8h blo = *(const v8h*)(wp + kbase);
            const v8h bhi = *(const v8h*)(wp + 16 + kbase);
            const v16h bb = __builtin_shufflevector(blo, bhi, 0,1,2,3,4,5,6,7,8,9,10,11,12,13,14,15);
            acc[t] = __builtin_amdgcn_wmma_f32_16x16x32_f16(
                false, a, false, bb, (short)0, acc[t], false, false);
        }
    }

    float* myLds = lds + wave * (16 * OUT_DIM);
    #pragma unroll
    for (int t = 0; t < 8; ++t) {
        float bv = bias[t * 16 + mlo];
        v8h hh;
        #pragma unroll
        for (int r = 0; r < 8; ++r) {
            float v = acc[t][r] + bv;
            hh[r] = (_Float16)v;
            myLds[(r + moff) * OUT_DIM + t * 16 + mlo] = v;
        }
        *(v8h*)(hT + (size_t)(b * OUT_DIM + t * 16 + mlo) * NNODES
                   + rt * 16 + moff) = hh;
    }
    __syncthreads();

    if (lane < 16) {
        int row = lane;
        float s1 = 0.f, s2 = 0.f;
        for (int n = 0; n < OUT_DIM; ++n) {
            float hv = myLds[row * OUT_DIM + n];
            s1 += hv * att_w[n];
            s2 += hv * att_w[OUT_DIM + n];
        }
        e1o[b * NNODES + rt * 16 + row] =
            __builtin_amdgcn_exp2f(-LOG2E * (s1 + att_b[0]));
        e2o[b * NNODES + rt * 16 + row] =
            __builtin_amdgcn_exp2f(-LOG2E * s2);
    }
}

// ---------------------------------------------------------------------------
// Kernel 2: out = (sigmoid * adj) @ h.
// Each wave owns a 32x128 output tile: 2 A tiles (rows) share one set of
// 8 B tiles per K chunk -> 16 WMMAs per 8KB of B, halving cache traffic
// per WMMA. B tiles are loaded into distinct registers FIRST, then the
// sigmoid/pack VALU work for both A tiles runs under the load shadow,
// then the 16-WMMA chain issues.
// sigmoid(s2_i+s1_j+ab) = 1 / (1 + e2[i]*e1[j])  (exps precomputed).
// ---------------------------------------------------------------------------
__global__ __launch_bounds__(32 * WAVES_PER_BLOCK)
void gat_out_kernel(const float*    __restrict__ adj,
                    const _Float16* __restrict__ hT,
                    const float*    __restrict__ e1,
                    const float*    __restrict__ e2,
                    float*          __restrict__ out) {
    const int lane  = threadIdx.x & 31;
    const int wave  = threadIdx.x >> 5;
    const int tile  = blockIdx.x * WAVES_PER_BLOCK + wave;  // 0..1023
    const int b     = tile >> 7;
    const int rt    = tile & (TILES32_PER_BATCH - 1);       // 32-row tiles
    const int mlo   = lane & 15;
    const int hi    = lane >> 4;
    const int kbase = hi * 8;
    const int kofs  = hi * 16;
    const int moff  = hi * 8;

    const int row0 = rt * 32 + mlo;        // rows for A tile 0
    const int row1 = row0 + 16;            // rows for A tile 1
    const float e2r0 = e2[b * NNODES + row0];
    const float e2r1 = e2[b * NNODES + row1];

    const float*    adjr0 = adj + (size_t)(b * NNODES + row0) * NNODES;
    const float*    adjr1 = adj + (size_t)(b * NNODES + row1) * NNODES;
    const float*    e1p   = e1 + b * NNODES;
    const _Float16* hTb   = hT + (size_t)b * OUT_DIM * NNODES;

    v8f acc0[8] = {};
    v8f acc1[8] = {};

    for (int kk = 0; kk < NNODES; kk += 32) {
        // ---- issue B-tile loads first (distinct regs -> overlap w/ VALU)
        v16h bb[8];
        #pragma unroll
        for (int t = 0; t < 8; ++t) {
            const _Float16* hp = hTb + (size_t)(t * 16 + mlo) * NNODES + kk;
            const v8h blo = *(const v8h*)(hp + kbase);
            const v8h bhi = *(const v8h*)(hp + 16 + kbase);
            bb[t] = __builtin_shufflevector(blo, bhi, 0,1,2,3,4,5,6,7,8,9,10,11,12,13,14,15);
        }

        // ---- adj / e1 streams (+ stream-ahead prefetch on adj rows)
        __builtin_prefetch(adjr0 + kk + 512, 0, 0);
        __builtin_prefetch(adjr1 + kk + 512, 0, 0);
        v8f ad0a = *(const v8f*)(adjr0 + kk + kbase);
        v8f ad0b = *(const v8f*)(adjr0 + kk + 16 + kbase);
        v8f ad1a = *(const v8f*)(adjr1 + kk + kbase);
        v8f ad1b = *(const v8f*)(adjr1 + kk + 16 + kbase);
        v8f e1a  = *(const v8f*)(e1p + kk + kbase);
        v8f e1b  = *(const v8f*)(e1p + kk + 16 + kbase);

        // ---- build both f16 A tiles: g = 1/(1 + e2r*e1[j]); a = g*adj
        v16h a0, a1;
        #pragma unroll
        for (int i = 0; i < 8; ++i) {
            float g0a = __builtin_amdgcn_rcpf(__builtin_fmaf(e2r0, e1a[i], 1.0f));
            float g0b = __builtin_amdgcn_rcpf(__builtin_fmaf(e2r0, e1b[i], 1.0f));
            float g1a = __builtin_amdgcn_rcpf(__builtin_fmaf(e2r1, e1a[i], 1.0f));
            float g1b = __builtin_amdgcn_rcpf(__builtin_fmaf(e2r1, e1b[i], 1.0f));
            a0[i]     = (_Float16)(g0a * ad0a[i]);
            a0[i + 8] = (_Float16)(g0b * ad0b[i]);
            a1[i]     = (_Float16)(g1a * ad1a[i]);
            a1[i + 8] = (_Float16)(g1b * ad1b[i]);
        }

        // ---- 16 WMMAs sharing the B tiles
        #pragma unroll
        for (int t = 0; t < 8; ++t) {
            acc0[t] = __builtin_amdgcn_wmma_f32_16x16x32_f16(
                false, a0, false, bb[t], (short)0, acc0[t], false, false);
            acc1[t] = __builtin_amdgcn_wmma_f32_16x16x32_f16(
                false, a1, false, bb[t], (short)0, acc1[t], false, false);
        }
    }

    float* orow0 = out + (size_t)(b * NNODES + rt * 32) * OUT_DIM;
    float* orow1 = orow0 + (size_t)16 * OUT_DIM;
    #pragma unroll
    for (int t = 0; t < 8; ++t) {
        #pragma unroll
        for (int r = 0; r < 8; ++r) {
            orow0[(size_t)(r + moff) * OUT_DIM + t * 16 + mlo] = acc0[t][r];
            orow1[(size_t)(r + moff) * OUT_DIM + t * 16 + mlo] = acc1[t][r];
        }
    }
}

// ---------------------------------------------------------------------------
// inputs (setup_inputs order): x, adj, W, b, att_w, att_b
// workspace: hT f16 (8 MB) | WT f16 (64 KB) | e1 (128 KB) | e2 (128 KB)
// ---------------------------------------------------------------------------
extern "C" void kernel_launch(void* const* d_in, const int* in_sizes, int n_in,
                              void* d_out, int out_size, void* d_ws, size_t ws_size,
                              hipStream_t stream) {
    const float* x     = (const float*)d_in[0];
    const float* adj   = (const float*)d_in[1];
    const float* W     = (const float*)d_in[2];
    const float* bias  = (const float*)d_in[3];
    const float* att_w = (const float*)d_in[4];
    const float* att_b = (const float*)d_in[5];
    float* out = (float*)d_out;

    char* ws = (char*)d_ws;
    _Float16* hT = (_Float16*)(ws);                                    // 8 MB
    _Float16* WT = (_Float16*)(ws + 8u * 1024u * 1024u);               // 64 KB
    float*    e1 = (float*)(ws + 8u * 1024u * 1024u + 64u * 1024u);    // 128 KB
    float*    e2 = (float*)(ws + 8u * 1024u * 1024u + 192u * 1024u);   // 128 KB

    wt_convert_kernel<<<(IN_DIM * OUT_DIM) / 256, 256, 0, stream>>>(W, WT);

    gat_h_kernel<<<(NBATCH * TILES16_PER_BATCH) / WAVES_PER_BLOCK,
                   32 * WAVES_PER_BLOCK, 0, stream>>>(
        x, WT, bias, att_w, att_b, hT, e1, e2);

    gat_out_kernel<<<(NBATCH * TILES32_PER_BATCH) / WAVES_PER_BLOCK,
                     32 * WAVES_PER_BLOCK, 0, stream>>>(
        adj, hT, e1, e2, out);
}
